// CrossModalAttention_37142877176539
// MI455X (gfx1250) — compile-verified, matrix operand order checked by reading, not run on hardware
//
#include <hip/hip_runtime.h>
#include <cstdint>

// ---------------------------------------------------------------------------
// CDNA5 (gfx1250) transformer block. All heavy math through
// v_wmma_f32_16x16x32_bf16; LDS staging via gfx1250 async global->LDS loads
// when the builtin exists (fallback: vmem load + ds store).
// ---------------------------------------------------------------------------

#define DEV __device__ __forceinline__

typedef __attribute__((ext_vector_type(16))) __bf16 v16bf;
typedef __attribute__((ext_vector_type(8)))  float  v8f;
typedef int v4i __attribute__((ext_vector_type(4)));
typedef __attribute__((address_space(1))) v4i* as1_v4i_p;   // global v4i*
typedef __attribute__((address_space(3))) v4i* as3_v4i_p;   // LDS v4i*

union FragAB { uint4 q[2]; v16bf v; };

DEV unsigned short f2b(float f) {              // f32 -> bf16, round-nearest-even
  unsigned int u = __float_as_uint(f);
  unsigned int r = (u + 0x7FFFu + ((u >> 16) & 1u)) >> 16;
  return (unsigned short)r;
}
DEV float b2f(unsigned short h) { return __uint_as_float(((unsigned int)h) << 16); }

#if __has_builtin(__builtin_amdgcn_global_load_async_to_lds_b128)
#define USE_ASYNC 1
#else
#define USE_ASYNC 0
#endif

DEV void stage16(const unsigned short* g, unsigned short* l) {
#if USE_ASYNC
  // Integer round-trip gives legal addrspace-typed pointers:
  //  - AS1: flat address == global address
  //  - AS3: low 32 bits of a flat LDS address are the LDS byte offset
  __builtin_amdgcn_global_load_async_to_lds_b128(
      (as1_v4i_p)(uintptr_t)g,
      (as3_v4i_p)(unsigned int)(uintptr_t)l, 0, 0);
#else
  *(uint4*)l = *(const uint4*)g;
#endif
}

template <int N> DEV void wait_async() {
#if USE_ASYNC
#if __has_builtin(__builtin_amdgcn_s_wait_asynccnt)
  __builtin_amdgcn_s_wait_asynccnt(N);
#else
  asm volatile("s_wait_asynccnt %0" ::"i"(N) : "memory");
#endif
#endif
}

enum { EPI_MOD = 0, EPI_QKV = 1, EPI_F32 = 2, EPI_GELU = 3, EPI_SCORES = 4, EPI_CTX = 5 };

// C[M,N] = A[M,K](bf16,row-major,ldA) x W[N,K](bf16,row-major) + bias[n]
// Block tile 128x128, 8 waves (wave32) each computing 32x64 via 16x16x32 WMMA.
template <int EPI>
__global__ __launch_bounds__(256) void gemm_bf16(
    const unsigned short* __restrict__ A, const unsigned short* __restrict__ W,
    const float* __restrict__ bias, const int* __restrict__ mod_ids,
    float* __restrict__ outF, unsigned short* __restrict__ oB0,
    unsigned short* __restrict__ oB1, unsigned short* __restrict__ oB2,
    int M, int N, int K, int ldA) {
  __shared__ __attribute__((aligned(16))) unsigned short As[2][128 * 40];
  __shared__ __attribute__((aligned(16))) unsigned short Ws[2][128 * 40];

  const int tid  = threadIdx.x;
  const int wave = tid >> 5;
  const int lane = tid & 31;
  const int r    = lane & 15;
  const int half = lane >> 4;
  const int wm   = wave >> 1;   // 0..3 -> 32-row slab
  const int wn   = wave & 1;    // 0..1 -> 64-col slab
  const int z    = blockIdx.z;
  const int m0   = blockIdx.y * 128;
  const int n0   = blockIdx.x * 128;

  const unsigned short* Ab = A;
  const unsigned short* Wb = W;
  const float* bs = bias;
  if constexpr (EPI == EPI_MOD) {        // per-position modality weight select
    int mid = mod_ids[z];
    Ab += (size_t)z * 1024;              // A row = batch index, s fixed = z
    Wb += (size_t)mid * (size_t)N * (size_t)K;
    bs += (size_t)mid * 1024;
  }
  if constexpr (EPI == EPI_SCORES || EPI == EPI_CTX) {  // z = b*H + h batch
    Ab += (size_t)z * 16384;
    Wb += (size_t)z * 16384;
  }
  Ab += (size_t)m0 * (size_t)ldA;
  Wb += (size_t)n0 * (size_t)K;
  const int ldW = K;

  auto stage_block = [&](int buf, int kc) {
    // 128 rows x 32 halfs = 512 x 16B chunks per matrix; 2 chunks/thread each
#pragma unroll
    for (int u = 0; u < 2; ++u) {
      int c = tid * 2 + u, row = c >> 2, sub = c & 3;
      stage16(Ab + (size_t)row * ldA + kc + sub * 8, &As[buf][row * 40 + sub * 8]);
    }
#pragma unroll
    for (int u = 0; u < 2; ++u) {
      int c = tid * 2 + u, row = c >> 2, sub = c & 3;
      stage16(Wb + (size_t)row * ldW + kc + sub * 8, &Ws[buf][row * 40 + sub * 8]);
    }
  };

  v8f acc[2][4] = {};

  stage_block(0, 0);
  const int nk = K >> 5;
  for (int s = 0; s < nk; ++s) {
    const int buf = s & 1;
    if (s + 1 < nk) { stage_block(buf ^ 1, (s + 1) << 5); wait_async<4>(); }
    else            { wait_async<0>(); }
    __syncthreads();

    FragAB af[2], wf[4];
#pragma unroll
    for (int i = 0; i < 2; ++i) {   // A frag: lane r = row, half selects K group
      const unsigned short* p = &As[buf][(wm * 32 + i * 16 + r) * 40 + half * 8];
      af[i].q[0] = *(const uint4*)p;
      af[i].q[1] = *(const uint4*)(p + 16);
    }
#pragma unroll
    for (int j = 0; j < 4; ++j) {   // B frag: lane r = output column
      const unsigned short* p = &Ws[buf][(wn * 64 + j * 16 + r) * 40 + half * 8];
      wf[j].q[0] = *(const uint4*)p;
      wf[j].q[1] = *(const uint4*)(p + 16);
    }
#pragma unroll
    for (int i = 0; i < 2; ++i)
#pragma unroll
      for (int j = 0; j < 4; ++j)
        acc[i][j] = __builtin_amdgcn_wmma_f32_16x16x32_bf16(
            false, af[i].v, false, wf[j].v, (short)0, acc[i][j], false, false);
    __syncthreads();
  }

  // Epilogue: C layout VGPR v -> row = base + half*8 + v, col = base + r
#pragma unroll
  for (int i = 0; i < 2; ++i) {
#pragma unroll
    for (int j = 0; j < 4; ++j) {
      const int rowBase = m0 + wm * 32 + i * 16 + half * 8;
      const int col     = n0 + wn * 64 + j * 16 + r;
      float bv = 0.f;
      if constexpr (EPI == EPI_MOD || EPI == EPI_QKV || EPI == EPI_F32 || EPI == EPI_GELU)
        bv = bs[col];
#pragma unroll
      for (int v = 0; v < 8; ++v) {
        const int row = rowBase + v;
        float val = acc[i][j][v] + bv;
        if constexpr (EPI == EPI_MOD) {
          size_t o = ((size_t)(row * 100 + z)) * 1024 + col;   // token=(b,s=z)
          outF[o] = val; oB0[o] = f2b(val);
        } else if constexpr (EPI == EPI_QKV) {
          int b = row / 100, sx = row - b * 100;
          int part = col >> 10, d = col & 1023, h = d >> 7, e = d & 127;
          size_t bh = (size_t)(b * 8 + h);
          if (part == 0)      oB0[(bh * 128 + sx) * 128 + e] = f2b(val);  // q
          else if (part == 1) oB1[(bh * 128 + sx) * 128 + e] = f2b(val);  // k
          else                oB2[(bh * 128 + e) * 128 + sx] = f2b(val);  // v^T
        } else if constexpr (EPI == EPI_F32) {
          outF[(size_t)row * N + col] = val;
        } else if constexpr (EPI == EPI_GELU) {
          float g = 0.5f * val * (1.f + erff(val * 0.70710678118654752f));
          oB0[(size_t)row * N + col] = f2b(g);
        } else if constexpr (EPI == EPI_SCORES) {
          outF[(size_t)z * 16384 + row * 128 + col] = val * 0.08838834764831845f;
        } else {  // EPI_CTX
          if (row < 100) {
            int b = z >> 3, h = z & 7;
            oB0[((size_t)(b * 100 + row)) * 1024 + h * 128 + col] = f2b(val);
          }
        }
      }
    }
  }
}

// ---- softmax over j<100 of a 128x128 scores tile, write bf16 padded attn ----
__global__ __launch_bounds__(256) void softmax_k(const float* __restrict__ sc,
                                                 unsigned short* __restrict__ attn) {
  int bh = blockIdx.x;
  int wave = threadIdx.x >> 5, lane = threadIdx.x & 31;
  int row = blockIdx.y * 8 + wave;
  if (row >= 100) return;
  const float* s = sc + (size_t)bh * 16384 + (size_t)row * 128;
  float v[4];
#pragma unroll
  for (int t = 0; t < 4; ++t) { int j = t * 32 + lane; v[t] = (j < 100) ? s[j] : -3.0e38f; }
  float mx = fmaxf(fmaxf(v[0], v[1]), fmaxf(v[2], v[3]));
#pragma unroll
  for (int o = 16; o; o >>= 1) mx = fmaxf(mx, __shfl_xor(mx, o));
  float sum = 0.f, p[4];
#pragma unroll
  for (int t = 0; t < 4; ++t) {
    int j = t * 32 + lane;
    p[t] = (j < 100) ? __expf(v[t] - mx) : 0.f;
    sum += p[t];
  }
#pragma unroll
  for (int o = 16; o; o >>= 1) sum += __shfl_xor(sum, o);
  float inv = 1.f / sum;
  unsigned short* a = attn + (size_t)bh * 16384 + (size_t)row * 128;
#pragma unroll
  for (int t = 0; t < 4; ++t) a[t * 32 + lane] = f2b(p[t] * inv);
}

// ---- attention weights: mean over heads -> d_out tail ----
__global__ void wmean_k(const unsigned short* __restrict__ attn, float* __restrict__ w) {
  int idx = blockIdx.x * 256 + threadIdx.x;
  if (idx >= 2560000) return;
  int b = idx / 10000, rem = idx - b * 10000;
  int i = rem / 100, j = rem - i * 100;
  float s = 0.f;
#pragma unroll
  for (int h = 0; h < 8; ++h) s += b2f(attn[(((size_t)(b * 8 + h)) * 128 + i) * 128 + j]);
  w[idx] = s * 0.125f;
}

// ---- LayerNorm(xa + xb) -> optional f32 + bf16, one token per block ----
__global__ __launch_bounds__(256) void ln_k(const float* __restrict__ xa,
    const float* __restrict__ xb, const float* __restrict__ g,
    const float* __restrict__ be, float* __restrict__ outF,
    unsigned short* __restrict__ outB) {
  __shared__ float red[16];
  const int tok = blockIdx.x;
  const int tid = threadIdx.x, wave = tid >> 5, lane = tid & 31;
  const float* pa = xa + (size_t)tok * 1024;
  const float* pb = xb + (size_t)tok * 1024;
  float vals[4], s = 0.f, s2 = 0.f;
#pragma unroll
  for (int t = 0; t < 4; ++t) {
    int e = t * 256 + tid;
    float v = pa[e] + pb[e];
    vals[t] = v; s += v; s2 += v * v;
  }
#pragma unroll
  for (int o = 16; o; o >>= 1) { s += __shfl_xor(s, o); s2 += __shfl_xor(s2, o); }
  if (lane == 0) { red[wave] = s; red[8 + wave] = s2; }
  __syncthreads();
  float S = 0.f, S2 = 0.f;
#pragma unroll
  for (int w = 0; w < 8; ++w) { S += red[w]; S2 += red[8 + w]; }
  float mean = S * (1.f / 1024.f);
  float var  = S2 * (1.f / 1024.f) - mean * mean;
  float rstd = rsqrtf(var + 1e-5f);
#pragma unroll
  for (int t = 0; t < 4; ++t) {
    int e = t * 256 + tid;
    float o = (vals[t] - mean) * rstd * g[e] + be[e];
    if (outF) outF[(size_t)tok * 1024 + e] = o;
    outB[(size_t)tok * 1024 + e] = f2b(o);
  }
}

// ---- small helpers ----
__global__ void prep_k(const float* __restrict__ emb, const float* __restrict__ pos,
                       unsigned short* __restrict__ xb, int n) {
  int base = blockIdx.x * 1024 + threadIdx.x;
#pragma unroll
  for (int t = 0; t < 4; ++t) {
    int idx = base + t * 256;
    if (idx < n) xb[idx] = f2b(emb[idx] + pos[idx % 102400]);
  }
}
__global__ void f2b_k(const float* __restrict__ s, unsigned short* __restrict__ d, int n) {
  int base = blockIdx.x * 1024 + threadIdx.x;
#pragma unroll
  for (int t = 0; t < 4; ++t) { int idx = base + t * 256; if (idx < n) d[idx] = f2b(s[idx]); }
}
__global__ void cbias_k(const float* __restrict__ mb, const float* __restrict__ me,
                        float* __restrict__ cb) {
  int i = blockIdx.x * 256 + threadIdx.x;
  if (i < 4096) cb[i] = mb[i] + me[i];
}
__global__ void fill0_k(unsigned short* __restrict__ d, int n) {
  int base = blockIdx.x * 1024 + threadIdx.x;
#pragma unroll
  for (int t = 0; t < 4; ++t) { int idx = base + t * 256; if (idx < n) d[idx] = 0; }
}

extern "C" void kernel_launch(void* const* d_in, const int* in_sizes, int n_in,
                              void* d_out, int out_size, void* d_ws, size_t ws_size,
                              hipStream_t stream) {
  const float* emb    = (const float*)d_in[0];
  const int*   modids = (const int*)  d_in[1];
  const float* pos    = (const float*)d_in[2];
  const float* modW   = (const float*)d_in[3];
  const float* modb   = (const float*)d_in[4];
  const float* modemb = (const float*)d_in[5];
  const float* inW    = (const float*)d_in[6];
  const float* inb    = (const float*)d_in[7];
  const float* outW   = (const float*)d_in[8];
  const float* outb   = (const float*)d_in[9];
  const float* w1     = (const float*)d_in[10];
  const float* b1     = (const float*)d_in[11];
  const float* w2     = (const float*)d_in[12];
  const float* b2     = (const float*)d_in[13];
  const float* ln1g   = (const float*)d_in[14];
  const float* ln1b   = (const float*)d_in[15];
  const float* ln2g   = (const float*)d_in[16];
  const float* ln2b   = (const float*)d_in[17];
  const float* poW    = (const float*)d_in[18];
  const float* pob    = (const float*)d_in[19];
  (void)in_sizes; (void)n_in; (void)out_size; (void)ws_size;

  char* ws = (char*)d_ws;
  auto U16 = [&](size_t off) { return (unsigned short*)(ws + off); };
  auto F32 = [&](size_t off) { return (float*)(ws + off); };

  // Lifetime-aliased workspace layout (~858 MB total)
  const size_t O_SLOTA = 0;                      // xin_b -> ctx_b -> x2_b (52MB)
  const size_t O_SLOTB = 52428800ULL;            // x_b -> x1_b (52MB)
  const size_t O_XF    = 104857600ULL;           // x after modality (f32)
  const size_t O_X1F   = 209715200ULL;           // x1 (f32)
  const size_t O_ATTF  = 314572800ULL;           // attended / ffn2 out (f32)
  const size_t O_ATTN  = 419430400ULL;           // attn bf16 [B,H,128,128]
  const size_t O_Q     = 486539264ULL;           // q [B,H,128,128] bf16
  const size_t O_K     = O_Q + 67108864ULL;      // k
  const size_t O_VT    = O_K + 67108864ULL;      // v^T [B,H,128,128]
  const size_t O_SC    = O_VT + 67108864ULL;     // scores f32 (134MB)
  const size_t O_H     = O_Q;                    // FFN hidden reuses q..scores
  const size_t O_WPOOL = O_SC + 134217728ULL;
  const size_t O_MODWB = O_WPOOL;
  const size_t O_INWB  = O_MODWB + 8388608ULL;
  const size_t O_OUTWB = O_INWB + 6291456ULL;
  const size_t O_W1B   = O_OUTWB + 2097152ULL;
  const size_t O_W2B   = O_W1B + 8388608ULL;
  const size_t O_POB   = O_W2B + 8388608ULL;
  const size_t O_CB    = O_POB + 2097152ULL;

  auto conv = [&](const float* s, size_t off, int n) {
    f2b_k<<<(n + 1023) / 1024, 256, 0, stream>>>(s, U16(off), n);
  };
  conv(modW, O_MODWB, 4 * 1024 * 1024);
  conv(inW,  O_INWB,  3072 * 1024);
  conv(outW, O_OUTWB, 1024 * 1024);
  conv(w1,   O_W1B,   4096 * 1024);
  conv(w2,   O_W2B,   4096 * 1024);
  conv(poW,  O_POB,   1024 * 1024);
  cbias_k<<<16, 256, 0, stream>>>(modb, modemb, F32(O_CB));
  prep_k<<<(26214400 + 1023) / 1024, 256, 0, stream>>>(emb, pos, U16(O_SLOTA), 26214400);
  fill0_k<<<98304, 256, 0, stream>>>(U16(O_Q), 100663296);   // zero q,k,v^T padding

  // modality dispatch: per-s weight, M=256(batch), N=K=1024
  gemm_bf16<EPI_MOD><<<dim3(8, 2, 100), 256, 0, stream>>>(
      U16(O_SLOTA), U16(O_MODWB), F32(O_CB), modids,
      F32(O_XF), U16(O_SLOTB), nullptr, nullptr, 256, 1024, 1024, 102400);
  // QKV projection with head-major scatter
  gemm_bf16<EPI_QKV><<<dim3(24, 200, 1), 256, 0, stream>>>(
      U16(O_SLOTB), U16(O_INWB), inb, nullptr,
      nullptr, U16(O_Q), U16(O_K), U16(O_VT), 25600, 3072, 1024, 1024);
  // scores = q k^T / sqrt(hd), batched over b*h
  gemm_bf16<EPI_SCORES><<<dim3(1, 1, 2048), 256, 0, stream>>>(
      U16(O_Q), U16(O_K), nullptr, nullptr,
      F32(O_SC), nullptr, nullptr, nullptr, 128, 128, 128, 128);
  softmax_k<<<dim3(2048, 13), 256, 0, stream>>>(F32(O_SC), U16(O_ATTN));
  wmean_k<<<10000, 256, 0, stream>>>(U16(O_ATTN), (float*)d_out + 26214400);
  // ctx = attn @ v, batched
  gemm_bf16<EPI_CTX><<<dim3(1, 1, 2048), 256, 0, stream>>>(
      U16(O_ATTN), U16(O_VT), nullptr, nullptr,
      nullptr, U16(O_SLOTA), nullptr, nullptr, 128, 128, 128, 128);
  // attended = ctx @ out_w^T + out_b
  gemm_bf16<EPI_F32><<<dim3(8, 200, 1), 256, 0, stream>>>(
      U16(O_SLOTA), U16(O_OUTWB), outb, nullptr,
      F32(O_ATTF), nullptr, nullptr, nullptr, 25600, 1024, 1024, 1024);
  // x1 = LN1(x + attended)
  ln_k<<<25600, 256, 0, stream>>>(F32(O_XF), F32(O_ATTF), ln1g, ln1b,
                                  F32(O_X1F), U16(O_SLOTB));
  // h = gelu(x1 @ w1^T + b1)
  gemm_bf16<EPI_GELU><<<dim3(32, 200, 1), 256, 0, stream>>>(
      U16(O_SLOTB), U16(O_W1B), b1, nullptr,
      nullptr, U16(O_H), nullptr, nullptr, 25600, 4096, 1024, 1024);
  // ff = h @ w2^T + b2
  gemm_bf16<EPI_F32><<<dim3(8, 200, 1), 256, 0, stream>>>(
      U16(O_H), U16(O_W2B), b2, nullptr,
      F32(O_ATTF), nullptr, nullptr, nullptr, 25600, 1024, 4096, 4096);
  // x2 = LN2(x1 + ff)
  ln_k<<<25600, 256, 0, stream>>>(F32(O_X1F), F32(O_ATTF), ln2g, ln2b,
                                  nullptr, U16(O_SLOTA));
  // out = x2 @ po_w^T + po_b  -> d_out
  gemm_bf16<EPI_F32><<<dim3(8, 200, 1), 256, 0, stream>>>(
      U16(O_SLOTA), U16(O_POB), pob, nullptr,
      (float*)d_out, nullptr, nullptr, nullptr, 25600, 1024, 1024, 1024);
}
